// GRU2D_55705725829833
// MI455X (gfx1250) — matrix operand order repaired and checked by reading, not compiled, not run on hardware
//
#include <hip/hip_runtime.h>
#include <hip/hip_bf16.h>
#include <stdint.h>

typedef __bf16 bf16;
typedef bf16  v16bf __attribute__((ext_vector_type(16)));
typedef float v8f   __attribute__((ext_vector_type(8)));
typedef float v4f   __attribute__((ext_vector_type(4)));
typedef unsigned int uint4v __attribute__((ext_vector_type(4)));

#define AS1 __attribute__((address_space(1)))
typedef const bf16  AS1* gbf16p;
typedef const uint4v AS1* gq4p;

union V16BF {
    struct { uint4v lo, hi; } q;   // two 16-byte chunks -> b128 loads
    v16bf v;
};

#define Bsz   512
#define Wd    256
#define TILEB 16
#define NWG   (Bsz / TILEB)   // 32 workgroups
#define NTHR  256             // 8 waves (wave32)

// ---- LDS layout (bytes), dynamic extern shared ----
#define OFF_ROWBUF   0                         // bf16 [16 cols][16 b][256]  131072
#define OFF_HLEFT    131072                    // bf16 [16][256]             8192
#define OFF_HINT_B   139264                    // bf16 [16][256]             8192
#define OFF_HINT_F   147456                    // f32  [16][256]             16384
#define OFF_GH       163840                    // f32  [16][768]             49152
#define OFF_GIC0     212992                    // f32  [768]                 3072
#define OFF_GIC1     216064                    // f32  [768]                 3072
#define OFF_WFC      219136                    // f32  [2][256]              2048
#define OFF_FCPART   221184                    // f32  [16][16][2]           2048
#define OFF_PUP      223232                    // f32  [16]
#define OFF_XLEFT    223296                    // f32  [16]
#define OFF_LP       223360                    // f32  [16]
#define OFF_BFC      223424                    // f32  [2]
#define SMEM_BYTES   223488

// Defeat LICM: launder the base address through an SGPR so weight-tile global
// loads stay inside the loop (L2-resident) instead of being hoisted across
// 256 iterations and spilled to scratch. Re-cast to address_space(1) so the
// loads lower to global_load_b128 (LOADcnt) rather than flat_load (LOADcnt+DScnt).
__device__ __forceinline__ gbf16p launder_global(const bf16* p) {
    uintptr_t v = (uintptr_t)p;
    asm volatile("" : "+s"(v));
    return (gbf16p)v;
}

__global__ void gru2d_cvt_weights(const float* __restrict__ W_hcat,
                                  const float* __restrict__ W_hh,
                                  bf16* __restrict__ whc_bf,
                                  bf16* __restrict__ whh_bf) {
    int i = blockIdx.x * blockDim.x + threadIdx.x;
    if (i < 256 * 512) whc_bf[i] = (bf16)W_hcat[i];
    if (i < 768 * 256) whh_bf[i] = (bf16)W_hh[i];
}

__device__ __forceinline__ v16bf load_a_tile(const bf16* base, int m, int half, int kt) {
    // 16-bit A 16x32: lane m (=lane&15), half (=lane>>4); K = kt*32 + half*16 + {0..15}
    // in elements 0..15 (contiguous, matching B).
    V16BF a;
    int off = m * Wd + kt * 32 + half * 16;
    a.q.lo = *(const uint4v*)(base + off);
    a.q.hi = *(const uint4v*)(base + off + 8);
    return a.v;
}

__device__ __forceinline__ v16bf load_b_tile(gbf16p wrow, int kb) {
    // 16-bit B 32x16: lane gives N col; lanes<16 hold K 0..15, lanes>=16 K 16..31.
    // wrow = &W[n][0]; 16 contiguous K values starting at kb.
    V16BF b;
    gq4p p = (gq4p)(wrow + kb);
    b.q.lo = p[0];
    b.q.hi = p[1];
    return b.v;
}

__device__ __forceinline__ float fast_sigmoid(float v) {
    return 1.0f / (1.0f + __expf(-v));
}
__device__ __forceinline__ float fast_tanh(float v) {
    return 2.0f / (1.0f + __expf(-2.0f * v)) - 1.0f;
}

__global__ __launch_bounds__(NTHR) void gru2d_persistent(
    const float* __restrict__ x,      // [512][16][16]
    const bf16*  __restrict__ Whc,    // [256][512] bf16 (W_hcat; l = cols 0..255, u = 256..511)
    const bf16*  __restrict__ Whh,    // [768][256] bf16
    const float* __restrict__ W_ih,   // [768][2]
    const float* __restrict__ b_ih,   // [768]
    const float* __restrict__ b_hh,   // [768]
    const float* __restrict__ W_fc,   // [2][256]
    const float* __restrict__ b_fc,   // [2]
    const float* __restrict__ W_xcat, // [2]
    float* __restrict__ out)          // [512]
{
    extern __shared__ char smem[];
    bf16*  rowbuf  = (bf16*) (smem + OFF_ROWBUF);
    bf16*  h_left  = (bf16*) (smem + OFF_HLEFT);
    bf16*  h_int_b = (bf16*) (smem + OFF_HINT_B);
    float* h_int_f = (float*)(smem + OFF_HINT_F);
    float* gh      = (float*)(smem + OFF_GH);
    float* gi_c0   = (float*)(smem + OFF_GIC0);
    float* gi_c1   = (float*)(smem + OFF_GIC1);
    float* wfc     = (float*)(smem + OFF_WFC);
    float* fcpart  = (float*)(smem + OFF_FCPART);
    float* p_up    = (float*)(smem + OFF_PUP);
    float* xleft   = (float*)(smem + OFF_XLEFT);
    float* lp_acc  = (float*)(smem + OFF_LP);
    float* bfc_s   = (float*)(smem + OFF_BFC);

    const int tid  = threadIdx.x;
    const int wg   = blockIdx.x;       // batch tile: rows wg*16 .. wg*16+15
    const int lane = tid & 31;
    const int wave = tid >> 5;

    // --- one-time LDS init ---
    {   // zero the 128KB row buffer with b128 stores
        uint4v z = {0u, 0u, 0u, 0u};
        uint4v* rb = (uint4v*)rowbuf;
        for (int i = tid; i < 131072 / 16; i += NTHR) rb[i] = z;
    }
    for (int g = tid; g < 768; g += NTHR) {
        float w0 = W_ih[2 * g], w1 = W_ih[2 * g + 1];
        gi_c0[g] = w1 + b_ih[g];      // gi = c0 + p * c1
        gi_c1[g] = w0 - w1;
    }
    for (int i = tid; i < 512; i += NTHR) wfc[i] = W_fc[i];
    if (tid < 16) lp_acc[tid] = 0.0f;
    if (tid < 2)  bfc_s[tid] = b_fc[tid];
    const float xc0 = W_xcat[0], xc1 = W_xcat[1];
    __syncthreads();

    const int a_m    = lane & 15;          // A row
    const int a_half = lane >> 4;          // A/B K-half select
    const int d_n    = lane & 15;          // C/D column within tile
    const int d_mb   = (lane >> 4) * 8;    // C/D first row

    for (int row = 0; row < 16; ++row) {
        // reset carries for this row
        {
            uint4v z = {0u, 0u, 0u, 0u};
            uint4v* hl = (uint4v*)h_left;
            for (int i = tid; i < 8192 / 16; i += NTHR) hl[i] = z;
        }
        if (tid < 16) xleft[tid] = 0.0f;
        __syncthreads();

        for (int t = 0; t < 16; ++t) {
            const int c = (row & 1) ? (15 - t) : t;       // canonical column
            const bf16* hup = rowbuf + c * (TILEB * Wd);  // previous row's h at column c

            if (tid < 16) {
                const int bg = wg * 16 + tid;
                float xu = (row > 0) ? x[bg * 256 + (row - 1) * 16 + c] : 0.0f;
                float xi = xleft[tid] * xc0 + xu * xc1;
                p_up[tid] = (xi + 1.0f) * 0.5f;
            }

            // ---- GEMM1: h_int = h_left @ Wh_l^T + h_up @ Wh_u^T ----
            v8f acc0 = {0.f,0.f,0.f,0.f,0.f,0.f,0.f,0.f};
            v8f acc1 = acc0;
            const int kbase = a_half * 16;
            for (int kt = 0; kt < 8; ++kt) {
                gbf16p whc_t = launder_global(Whc);   // keep B loads inside the loop
                v16bf aL = load_a_tile(h_left, a_m, a_half, kt);
                v16bf aU = load_a_tile(hup,    a_m, a_half, kt);
                const int kb = kt * 32 + kbase;
                #pragma unroll
                for (int nt = 0; nt < 2; ++nt) {
                    const int ncol = (wave * 2 + nt) * 16 + d_n;
                    v16bf bL = load_b_tile(whc_t + ncol * 512,       kb);  // Wh_l
                    v16bf bU = load_b_tile(whc_t + ncol * 512 + 256, kb);  // Wh_u
                    v8f& acc = nt ? acc1 : acc0;
                    acc = __builtin_amdgcn_wmma_f32_16x16x32_bf16(
                              false, aL, false, bL, (short)0, acc, false, false);
                    acc = __builtin_amdgcn_wmma_f32_16x16x32_bf16(
                              false, aU, false, bU, (short)0, acc, false, false);
                }
            }
            #pragma unroll
            for (int nt = 0; nt < 2; ++nt) {
                const int ncol = (wave * 2 + nt) * 16 + d_n;
                v8f acc = nt ? acc1 : acc0;
                #pragma unroll
                for (int r = 0; r < 8; ++r) {
                    const int m = d_mb + r;
                    h_int_f[m * Wd + ncol] = acc[r];
                    h_int_b[m * Wd + ncol] = (bf16)acc[r];
                }
            }
            __syncthreads();

            // ---- GEMM2: gh = h_int @ W_hh^T + b_hh ----
            v8f gacc[6];
            {
                v8f z = {0.f,0.f,0.f,0.f,0.f,0.f,0.f,0.f};
                #pragma unroll
                for (int nt = 0; nt < 6; ++nt) gacc[nt] = z;
            }
            for (int kt = 0; kt < 8; ++kt) {
                gbf16p whh_t = launder_global(Whh);   // keep B loads inside the loop
                v16bf aI = load_a_tile(h_int_b, a_m, a_half, kt);
                const int kb = kt * 32 + kbase;
                #pragma unroll
                for (int nt = 0; nt < 6; ++nt) {
                    const int ncol = (wave * 6 + nt) * 16 + d_n;
                    v16bf bh = load_b_tile(whh_t + ncol * 256, kb);
                    gacc[nt] = __builtin_amdgcn_wmma_f32_16x16x32_bf16(
                                   false, aI, false, bh, (short)0, gacc[nt], false, false);
                }
            }
            #pragma unroll
            for (int nt = 0; nt < 6; ++nt) {
                const int ncol = (wave * 6 + nt) * 16 + d_n;
                const float bias = b_hh[ncol];
                #pragma unroll
                for (int r = 0; r < 8; ++r)
                    gh[(d_mb + r) * 768 + ncol] = gacc[nt][r] + bias;
            }
            __syncthreads();

            // ---- gates, h_next, fc partial dot products (vectorized LDS) ----
            {
                const int b  = tid >> 4;   // batch row 0..15
                const int hb = tid & 15;   // hidden block of 16
                const int j0 = hb * 16;
                const float p = p_up[b];

                const v4f* pc0r = (const v4f*)(gi_c0 + j0);
                const v4f* pc0z = (const v4f*)(gi_c0 + 256 + j0);
                const v4f* pc0n = (const v4f*)(gi_c0 + 512 + j0);
                const v4f* pc1r = (const v4f*)(gi_c1 + j0);
                const v4f* pc1z = (const v4f*)(gi_c1 + 256 + j0);
                const v4f* pc1n = (const v4f*)(gi_c1 + 512 + j0);
                const v4f* pghr = (const v4f*)(gh + b * 768 + j0);
                const v4f* pghz = (const v4f*)(gh + b * 768 + 256 + j0);
                const v4f* pghn = (const v4f*)(gh + b * 768 + 512 + j0);
                const v4f* phin = (const v4f*)(h_int_f + b * Wd + j0);
                const v4f* pw0  = (const v4f*)(wfc + j0);
                const v4f* pw1  = (const v4f*)(wfc + 256 + j0);

                union { bf16 h[16]; uint4v q[2]; } hpack;
                float s0 = 0.0f, s1 = 0.0f;
                #pragma unroll
                for (int q4 = 0; q4 < 4; ++q4) {
                    v4f c0r = pc0r[q4], c0z = pc0z[q4], c0n = pc0n[q4];
                    v4f c1r = pc1r[q4], c1z = pc1z[q4], c1n = pc1n[q4];
                    v4f ghr = pghr[q4], ghz = pghz[q4], ghn = pghn[q4];
                    v4f hin = phin[q4];
                    v4f w0  = pw0[q4],  w1  = pw1[q4];
                    #pragma unroll
                    for (int e = 0; e < 4; ++e) {
                        const float gir = c0r[e] + p * c1r[e];
                        const float giz = c0z[e] + p * c1z[e];
                        const float gin = c0n[e] + p * c1n[e];
                        const float r = fast_sigmoid(gir + ghr[e]);
                        const float z = fast_sigmoid(giz + ghz[e]);
                        const float n = fast_tanh(gin + r * ghn[e]);
                        const float hn = (1.0f - z) * n + z * hin[e];
                        hpack.h[q4 * 4 + e] = (bf16)hn;
                        s0 += hn * w0[e];
                        s1 += hn * w1[e];
                    }
                }
                uint4v* dstL = (uint4v*)(h_left + b * Wd + j0);
                dstL[0] = hpack.q[0];
                dstL[1] = hpack.q[1];
                uint4v* dstR = (uint4v*)(rowbuf + c * (TILEB * Wd) + b * Wd + j0);
                dstR[0] = hpack.q[0];
                dstR[1] = hpack.q[1];
                fcpart[(b * 16 + hb) * 2 + 0] = s0;
                fcpart[(b * 16 + hb) * 2 + 1] = s1;
            }
            __syncthreads();

            // ---- fc reduce, log_softmax, lp accumulation, x carry ----
            if (tid < 16) {
                const int b = tid;
                const v4f* fp = (const v4f*)(fcpart + b * 32);
                float y0 = bfc_s[0], y1 = bfc_s[1];
                #pragma unroll
                for (int i = 0; i < 8; ++i) {
                    v4f v = fp[i];
                    y0 += v[0] + v[2];
                    y1 += v[1] + v[3];
                }
                const float mx  = fmaxf(y0, y1);
                const float lse = mx + __logf(__expf(y0 - mx) + __expf(y1 - mx));
                const float ls0 = y0 - lse, ls1 = y1 - lse;
                const int bg = wg * 16 + b;
                const float xv = x[bg * 256 + row * 16 + c];
                const float m  = (1.0f + xv) * 0.5f;
                const float lp = (row == 0 && c == 0) ? (1.0f - m)
                                                      : (ls0 * m + ls1 * (1.0f - m));
                lp_acc[b] += lp;
                xleft[b] = xv;
            }
            // no end-of-step barrier needed: fcpart/p_up reuse is separated by
            // the two GEMM barriers of the next step; xleft/lp_acc are
            // same-thread carries.
        } // t (columns)
    } // row

    __syncthreads();
    if (tid < 16) out[wg * 16 + tid] = lp_acc[tid];
}

extern "C" void kernel_launch(void* const* d_in, const int* in_sizes, int n_in,
                              void* d_out, int out_size, void* d_ws, size_t ws_size,
                              hipStream_t stream) {
    const float* x      = (const float*)d_in[0];
    const float* W_ih   = (const float*)d_in[1];
    const float* W_hh   = (const float*)d_in[2];
    const float* b_ih   = (const float*)d_in[3];
    const float* b_hh   = (const float*)d_in[4];
    const float* W_fc   = (const float*)d_in[5];
    const float* b_fc   = (const float*)d_in[6];
    const float* W_hcat = (const float*)d_in[7];
    const float* W_xcat = (const float*)d_in[8];
    float* out = (float*)d_out;

    // workspace: bf16 weight copies (Whcat 256x512, Whh 768x256)
    bf16* whc_bf = (bf16*)d_ws;
    bf16* whh_bf = (bf16*)((char*)d_ws + (size_t)(256 * 512) * sizeof(bf16));

    gru2d_cvt_weights<<<768, 256, 0, stream>>>(W_hcat, W_hh, whc_bf, whh_bf);

    hipFuncSetAttribute((const void*)gru2d_persistent,
                        hipFuncAttributeMaxDynamicSharedMemorySize, SMEM_BYTES);

    gru2d_persistent<<<NWG, NTHR, SMEM_BYTES, stream>>>(
        x, whc_bf, whh_bf, W_ih, b_ih, b_hh, W_fc, b_fc, W_xcat, out);
}
